// BindingAffinityModel_16458314678575
// MI455X (gfx1250) — compile-verified, matrix operand order checked by reading, not run on hardware
//
#include <hip/hip_runtime.h>
#include <hip/hip_bf16.h>
#include <math.h>

// ---------------------------------------------------------------------------
// EGNN forward for MI455X (gfx1250, wave32, WMMA 16x16x32 f16 -> f32)
// ---------------------------------------------------------------------------

typedef _Float16 v8h  __attribute__((ext_vector_type(8)));
typedef _Float16 v16h __attribute__((ext_vector_type(16)));
typedef float    v8f  __attribute__((ext_vector_type(8)));

#define N_NODES 50000
#define N_EDGES 1600000
#define HID 32
#define EDIM 16
#define NLAYERS 4
#define FIN 23
#define WAVES_PER_BLOCK 8

// Weight-fragment table (each fragment: 32 lanes x 16 halfs = 512 halfs)
#define FRAG_PROJ(t)      (t)
#define FRAG_PE1(l,c,t)   (2  + ((l)*3+(c))*2 + (t))
#define FRAG_PE2(l,t)     (26 + (l)*2 + (t))
#define FRAG_PH1(l,c,t)   (34 + ((l)*2+(c))*2 + (t))
#define FRAG_PH2(l,t)     (50 + (l)*2 + (t))
#define FRAG_PX1(l,t)     (58 + (l)*2 + (t))
#define N_FRAGS 66

#define WS_FRAG_BYTES 131072  // fragments live at ws[0..128KB)

__device__ __forceinline__ float siluf(float v) {
  return v / (1.0f + __expf(-v));
}

__device__ __forceinline__ v8f zero8() {
  v8f z = {0.f,0.f,0.f,0.f,0.f,0.f,0.f,0.f};
  return z;
}

__device__ __forceinline__ v8f wmma16(v16h a, v16h b, v8f c) {
  // D = A(16x32 f16) * B(32x16 f16) + C(16x16 f32)
  return __builtin_amdgcn_wmma_f32_16x16x32_f16(false, a, false, b,
                                                (short)0, c, false, false);
}

// A-fragment for 16-bit 16x32: lane L holds row m = L&15; K halves
// kb..kb+7 (lo) and kb+16..kb+23 (hi) with kb = 8*(L>>4). rowBase must point
// at &stage[m][kOff + kb]; both 16B loads are aligned by construction.
__device__ __forceinline__ v16h load_afrag(const _Float16* rowBase) {
  v8h lo = *(const v8h*)(rowBase);
  v8h hi = *(const v8h*)(rowBase + 16);
  v16h a;
#pragma unroll
  for (int j = 0; j < 8; ++j) { a[j] = lo[j]; a[8 + j] = hi[j]; }
  return a;
}

__device__ __forceinline__ v16h load_wfrag(const _Float16* wf, int frag, int lane) {
  return *(const v16h*)(wf + (size_t)frag * 512 + (size_t)lane * 16);
}

__device__ __forceinline__ void atomicAddF32(float* p, float v) {
  __hip_atomic_fetch_add(p, v, __ATOMIC_RELAXED, __HIP_MEMORY_SCOPE_AGENT);
}

// ---------------------------------------------------------------------------
// Weight preparation: f32 (K x 32 row-major) -> f16 WMMA B fragments.
// B layout (32x16, 16-bit): lane L holds col n = L&15; K halves kb..kb+7 and
// kb+16..kb+23 with kb = 8*(L>>4)  (mirror of the A layout with M<->N).
// ---------------------------------------------------------------------------
__global__ void prep_weights_kernel(const float* __restrict__ proj_w,
                                    const float* __restrict__ pe_w1,
                                    const float* __restrict__ pe_w2,
                                    const float* __restrict__ ph_w1,
                                    const float* __restrict__ ph_w2,
                                    const float* __restrict__ px_w1,
                                    _Float16* __restrict__ wf) {
  int gid = blockIdx.x * blockDim.x + threadIdx.x;
  if (gid >= N_FRAGS * 512) return;
  int frag = gid >> 9;
  int lane = (gid >> 4) & 31;
  int j    = gid & 15;

  int n_local = lane & 15;
  int kb = (lane >> 4) * 8;
  int k_in_chunk = ((j >> 3) << 4) + kb + (j & 7);  // 0..31

  float val = 0.0f;
  if (frag < 2) {                       // proj_w: (23,32), K pad 23->32
    int t = frag;
    int k = k_in_chunk;
    if (k < FIN) val = proj_w[k * HID + t * 16 + n_local];
  } else if (frag < 26) {               // pe_w1: (L,81,32), K pad 81->96
    int f = frag - 2; int l = f / 6; int rem = f % 6; int c = rem >> 1; int t = rem & 1;
    int k = c * 32 + k_in_chunk;
    if (k < 2 * HID + EDIM + 1) val = pe_w1[((size_t)l * 81 + k) * HID + t * 16 + n_local];
  } else if (frag < 34) {               // pe_w2: (L,32,32)
    int f = frag - 26; int l = f >> 1; int t = f & 1;
    val = pe_w2[((size_t)l * HID + k_in_chunk) * HID + t * 16 + n_local];
  } else if (frag < 50) {               // ph_w1: (L,64,32)
    int f = frag - 34; int l = f >> 2; int rem = f & 3; int c = rem >> 1; int t = rem & 1;
    int k = c * 32 + k_in_chunk;
    val = ph_w1[((size_t)l * 64 + k) * HID + t * 16 + n_local];
  } else if (frag < 58) {               // ph_w2: (L,32,32)
    int f = frag - 50; int l = f >> 1; int t = f & 1;
    val = ph_w2[((size_t)l * HID + k_in_chunk) * HID + t * 16 + n_local];
  } else {                              // px_w1: (L,32,32)
    int f = frag - 58; int l = f >> 1; int t = f & 1;
    val = px_w1[((size_t)l * HID + k_in_chunk) * HID + t * 16 + n_local];
  }
  wf[gid] = (_Float16)val;
}

// ---------------------------------------------------------------------------
// Projection: h = node_feat @ proj_w + proj_b ; x = node_pos
// ---------------------------------------------------------------------------
struct alignas(16) ProjSmem {
  _Float16 stage[16][32];   // row = 64 B, aligned
};

__global__ void proj_kernel(const float* __restrict__ node_feat,
                            const float* __restrict__ node_pos,
                            const float* __restrict__ proj_b,
                            const _Float16* __restrict__ wf,
                            float* __restrict__ h, float* __restrict__ x) {
  __shared__ ProjSmem sm[WAVES_PER_BLOCK];
  int wave = threadIdx.x >> 5, lane = threadIdx.x & 31;
  int tile = blockIdx.x * WAVES_PER_BLOCK + wave;
  if (tile * 16 >= N_NODES) return;
  ProjSmem& s = sm[wave];

  int e = lane & 15, p = lane >> 4;
  int node = tile * 16 + e;
  const float* nf = node_feat + (size_t)node * FIN;
#pragma unroll
  for (int j = 0; j < 16; ++j) {
    int f = p * 16 + j;
    s.stage[e][f] = (f < FIN) ? (_Float16)nf[f] : (_Float16)0.0f;
  }
  __builtin_amdgcn_wave_barrier();

  int m = e, kb = p * 8;
  v16h a = load_afrag(&s.stage[m][kb]);
  v8f acc0 = zero8(), acc1 = zero8();
  acc0 = wmma16(a, load_wfrag(wf, FRAG_PROJ(0), lane), acc0);
  acc1 = wmma16(a, load_wfrag(wf, FRAG_PROJ(1), lane), acc1);

  int n0 = lane & 15, n1 = 16 + n0;
  float b0 = proj_b[n0], b1 = proj_b[n1];
  int mrow = p * 8;
#pragma unroll
  for (int r = 0; r < 8; ++r) {
    int nm = tile * 16 + mrow + r;
    h[(size_t)nm * HID + n0] = acc0[r] + b0;
    h[(size_t)nm * HID + n1] = acc1[r] + b1;
  }
  if (lane < 16) {
    int nm = tile * 16 + lane;
    x[nm * 3 + 0] = node_pos[nm * 3 + 0];
    x[nm * 3 + 1] = node_pos[nm * 3 + 1];
    x[nm * 3 + 2] = node_pos[nm * 3 + 2];
  }
}

// ---------------------------------------------------------------------------
// Edge kernel: one wave handles 16 edges. phi_e + agg scatter + phi_x + dx.
// ---------------------------------------------------------------------------
struct alignas(16) EdgeSmem {
  _Float16 stage[16][96];   // m_in staging, row = 192 B
  _Float16 act[16][32];     // inter-GEMM activations, row = 64 B
  float    fred[16][16];    // phi_x row reduction
  float    diff3[16][3];
  float    invn[16];
  int      dstIdx[16];
};

__global__ void egnn_edge_kernel(const float* __restrict__ h,
                                 const float* __restrict__ x,
                                 const float* __restrict__ edge_attr,
                                 const int* __restrict__ eidx,
                                 const float* __restrict__ pe_b1,
                                 const float* __restrict__ pe_b2,
                                 const float* __restrict__ px_b1,
                                 const float* __restrict__ px_w2,
                                 const float* __restrict__ px_b2,
                                 const _Float16* __restrict__ wf,
                                 float* __restrict__ agg,
                                 float* __restrict__ dxbuf,
                                 int layer) {
  __shared__ EdgeSmem sm[WAVES_PER_BLOCK];
  int wave = threadIdx.x >> 5, lane = threadIdx.x & 31;
  int tile = blockIdx.x * WAVES_PER_BLOCK + wave;
  if (tile * 16 >= N_EDGES) return;
  EdgeSmem& s = sm[wave];

  // Warm the weight path in GL2 (global_prefetch_b8).
  __builtin_prefetch(wf + (size_t)FRAG_PE1(layer, 0, 0) * 512, 0, 0);

  int e = lane & 15, p = lane >> 4;
  int ge = tile * 16 + e;
  int si = eidx[ge], di = eidx[N_EDGES + ge];

  if (p == 0) {
    const float* hr = h + (size_t)si * HID;
#pragma unroll
    for (int j = 0; j < HID; ++j) s.stage[e][j] = (_Float16)hr[j];
    const float* ea = edge_attr + (size_t)ge * EDIM;
#pragma unroll
    for (int j = 0; j < EDIM; ++j) s.stage[e][2 * HID + j] = (_Float16)ea[j];
    s.dstIdx[e] = di;
  } else {
    const float* hr = h + (size_t)di * HID;
#pragma unroll
    for (int j = 0; j < HID; ++j) s.stage[e][HID + j] = (_Float16)hr[j];
    float d0 = x[di * 3 + 0] - x[si * 3 + 0];
    float d1 = x[di * 3 + 1] - x[si * 3 + 1];
    float d2 = x[di * 3 + 2] - x[si * 3 + 2];
    float dd = d0 * d0 + d1 * d1 + d2 * d2;
    s.stage[e][80] = (_Float16)sqrtf(dd + 1e-9f);
#pragma unroll
    for (int j = 81; j < 96; ++j) s.stage[e][j] = (_Float16)0.0f;
    s.diff3[e][0] = d0; s.diff3[e][1] = d1; s.diff3[e][2] = d2;
    s.invn[e] = 1.0f / (sqrtf(dd) + 1e-9f);
  }
  __builtin_amdgcn_wave_barrier();

  int m = e, kb = p * 8;
  int n0 = lane & 15, n1 = 16 + n0;
  int mrow = p * 8;

  // ---- phi_e GEMM1: (16x96) @ (96x32), 3 K-chunks x 2 N-tiles ----
  v8f z0 = zero8(), z1 = zero8();
#pragma unroll
  for (int c = 0; c < 3; ++c) {
    v16h a = load_afrag(&s.stage[m][c * 32 + kb]);
    z0 = wmma16(a, load_wfrag(wf, FRAG_PE1(layer, c, 0), lane), z0);
    z1 = wmma16(a, load_wfrag(wf, FRAG_PE1(layer, c, 1), lane), z1);
  }
  float b10 = pe_b1[layer * HID + n0], b11 = pe_b1[layer * HID + n1];
#pragma unroll
  for (int r = 0; r < 8; ++r) {
    s.act[mrow + r][n0] = (_Float16)siluf(z0[r] + b10);
    s.act[mrow + r][n1] = (_Float16)siluf(z1[r] + b11);
  }
  __builtin_amdgcn_wave_barrier();

  // ---- phi_e GEMM2: (16x32) @ (32x32) -> m_ij ----
  v16h a2 = load_afrag(&s.act[m][kb]);
  v8f m0 = zero8(), m1 = zero8();
  m0 = wmma16(a2, load_wfrag(wf, FRAG_PE2(layer, 0), lane), m0);
  m1 = wmma16(a2, load_wfrag(wf, FRAG_PE2(layer, 1), lane), m1);
  float b20 = pe_b2[layer * HID + n0], b21 = pe_b2[layer * HID + n1];
#pragma unroll
  for (int r = 0; r < 8; ++r) { m0[r] += b20; m1[r] += b21; }

  // ---- scatter m_ij into agg (segment_sum over dst) + restage for phi_x ----
#pragma unroll
  for (int r = 0; r < 8; ++r) {
    int mm = mrow + r;
    int d  = s.dstIdx[mm];
    atomicAddF32(&agg[(size_t)d * HID + n0], m0[r]);
    atomicAddF32(&agg[(size_t)d * HID + n1], m1[r]);
    s.act[mm][n0] = (_Float16)m0[r];
    s.act[mm][n1] = (_Float16)m1[r];
  }
  __builtin_amdgcn_wave_barrier();

  // ---- phi_x GEMM: (16x32) @ (32x32), SiLU, dot with px_w2 ----
  v16h a3 = load_afrag(&s.act[m][kb]);
  v8f g0 = zero8(), g1 = zero8();
  g0 = wmma16(a3, load_wfrag(wf, FRAG_PX1(layer, 0), lane), g0);
  g1 = wmma16(a3, load_wfrag(wf, FRAG_PX1(layer, 1), lane), g1);
  float pb0 = px_b1[layer * HID + n0], pb1 = px_b1[layer * HID + n1];
  float w20 = px_w2[layer * HID + n0], w21 = px_w2[layer * HID + n1];
#pragma unroll
  for (int r = 0; r < 8; ++r) {
    s.fred[mrow + r][n0] = siluf(g0[r] + pb0) * w20 + siluf(g1[r] + pb1) * w21;
  }
  __builtin_amdgcn_wave_barrier();

  if (lane < 16) {
    float gate = px_b2[layer];
#pragma unroll
    for (int c = 0; c < 16; ++c) gate += s.fred[lane][c];
    int d = s.dstIdx[lane];
    float sc = s.invn[lane] * gate;
    atomicAddF32(&dxbuf[d * 3 + 0], s.diff3[lane][0] * sc);
    atomicAddF32(&dxbuf[d * 3 + 1], s.diff3[lane][1] * sc);
    atomicAddF32(&dxbuf[d * 3 + 2], s.diff3[lane][2] * sc);
  }
}

// ---------------------------------------------------------------------------
// Node kernel: phi_h([h, agg]) + residual + LayerNorm, x += dx.
// ---------------------------------------------------------------------------
struct alignas(16) NodeSmem {
  _Float16 stage[16][64];   // [h | agg], row = 128 B
  _Float16 act[16][32];
  float    fst[16][32];     // pre-LN rows
};

__global__ void egnn_node_kernel(const float* __restrict__ agg,
                                 const float* __restrict__ dxbuf,
                                 const float* __restrict__ ph_b1,
                                 const float* __restrict__ ph_b2,
                                 const float* __restrict__ ln_g,
                                 const float* __restrict__ ln_b,
                                 const _Float16* __restrict__ wf,
                                 float* __restrict__ h,
                                 float* __restrict__ x,
                                 int layer) {
  __shared__ NodeSmem sm[WAVES_PER_BLOCK];
  int wave = threadIdx.x >> 5, lane = threadIdx.x & 31;
  int tile = blockIdx.x * WAVES_PER_BLOCK + wave;
  if (tile * 16 >= N_NODES) return;
  NodeSmem& s = sm[wave];

  int e = lane & 15, p = lane >> 4;
  int node = tile * 16 + e;
  const float* src = (p == 0) ? (h + (size_t)node * HID) : (agg + (size_t)node * HID);
#pragma unroll
  for (int j = 0; j < HID; ++j) s.stage[e][p * HID + j] = (_Float16)src[j];
  __builtin_amdgcn_wave_barrier();

  int m = e, kb = p * 8;
  int n0 = lane & 15, n1 = 16 + n0;
  int mrow = p * 8;

  // GEMM1: (16x64) @ (64x32), 2 K-chunks x 2 N-tiles
  v8f z0 = zero8(), z1 = zero8();
#pragma unroll
  for (int c = 0; c < 2; ++c) {
    v16h a = load_afrag(&s.stage[m][c * 32 + kb]);
    z0 = wmma16(a, load_wfrag(wf, FRAG_PH1(layer, c, 0), lane), z0);
    z1 = wmma16(a, load_wfrag(wf, FRAG_PH1(layer, c, 1), lane), z1);
  }
  float b10 = ph_b1[layer * HID + n0], b11 = ph_b1[layer * HID + n1];
#pragma unroll
  for (int r = 0; r < 8; ++r) {
    s.act[mrow + r][n0] = (_Float16)siluf(z0[r] + b10);
    s.act[mrow + r][n1] = (_Float16)siluf(z1[r] + b11);
  }
  __builtin_amdgcn_wave_barrier();

  // GEMM2: (16x32) @ (32x32) -> dh
  v16h a2 = load_afrag(&s.act[m][kb]);
  v8f d0 = zero8(), d1 = zero8();
  d0 = wmma16(a2, load_wfrag(wf, FRAG_PH2(layer, 0), lane), d0);
  d1 = wmma16(a2, load_wfrag(wf, FRAG_PH2(layer, 1), lane), d1);
  float b20 = ph_b2[layer * HID + n0], b21 = ph_b2[layer * HID + n1];

  // residual rows -> LDS for LayerNorm
#pragma unroll
  for (int r = 0; r < 8; ++r) {
    int mm = mrow + r;
    int nm = tile * 16 + mm;
    s.fst[mm][n0] = h[(size_t)nm * HID + n0] + d0[r] + b20;
    s.fst[mm][n1] = h[(size_t)nm * HID + n1] + d1[r] + b21;
  }
  __builtin_amdgcn_wave_barrier();

  if (lane < 16) {
    int row = lane;
    int nm = tile * 16 + row;
    float mu = 0.0f;
#pragma unroll
    for (int c = 0; c < HID; ++c) mu += s.fst[row][c];
    mu *= (1.0f / HID);
    float var = 0.0f;
#pragma unroll
    for (int c = 0; c < HID; ++c) {
      float t = s.fst[row][c] - mu;
      var += t * t;
    }
    var *= (1.0f / HID);
    float inv = rsqrtf(var + 1e-5f);
#pragma unroll
    for (int c = 0; c < HID; ++c) {
      h[(size_t)nm * HID + c] =
          (s.fst[row][c] - mu) * inv * ln_g[layer * HID + c] + ln_b[layer * HID + c];
    }
    x[nm * 3 + 0] += dxbuf[nm * 3 + 0];
    x[nm * 3 + 1] += dxbuf[nm * 3 + 1];
    x[nm * 3 + 2] += dxbuf[nm * 3 + 2];
  }
}

// ---------------------------------------------------------------------------
extern "C" void kernel_launch(void* const* d_in, const int* in_sizes, int n_in,
                              void* d_out, int out_size, void* d_ws, size_t ws_size,
                              hipStream_t stream) {
  const float* node_pos  = (const float*)d_in[0];
  const float* node_feat = (const float*)d_in[1];
  const float* edge_attr = (const float*)d_in[2];
  const float* proj_w    = (const float*)d_in[3];
  const float* proj_b    = (const float*)d_in[4];
  const float* pe_w1     = (const float*)d_in[5];
  const float* pe_b1     = (const float*)d_in[6];
  const float* pe_w2     = (const float*)d_in[7];
  const float* pe_b2     = (const float*)d_in[8];
  const float* ph_w1     = (const float*)d_in[9];
  const float* ph_b1     = (const float*)d_in[10];
  const float* ph_w2     = (const float*)d_in[11];
  const float* ph_b2     = (const float*)d_in[12];
  const float* px_w1     = (const float*)d_in[13];
  const float* px_b1     = (const float*)d_in[14];
  const float* px_w2     = (const float*)d_in[15];
  const float* px_b2     = (const float*)d_in[16];
  const float* ln_g      = (const float*)d_in[17];
  const float* ln_b      = (const float*)d_in[18];
  const int*   eidx      = (const int*)d_in[19];

  float* h_out = (float*)d_out;                    // (N, 32)
  float* x_out = h_out + (size_t)N_NODES * HID;    // (N, 3)

  _Float16* wf  = (_Float16*)d_ws;
  float*    agg = (float*)((char*)d_ws + WS_FRAG_BYTES);
  float*    dxb = agg + (size_t)N_NODES * HID;

  prep_weights_kernel<<<(N_FRAGS * 512 + 255) / 256, 256, 0, stream>>>(
      proj_w, pe_w1, pe_w2, ph_w1, ph_w2, px_w1, wf);

  int nodeBlocks = (N_NODES / 16 + WAVES_PER_BLOCK - 1) / WAVES_PER_BLOCK;  // 391
  int edgeBlocks = (N_EDGES / 16 + WAVES_PER_BLOCK - 1) / WAVES_PER_BLOCK;  // 12500

  proj_kernel<<<nodeBlocks, 32 * WAVES_PER_BLOCK, 0, stream>>>(
      node_feat, node_pos, proj_b, wf, h_out, x_out);

  for (int l = 0; l < NLAYERS; ++l) {
    hipMemsetAsync(agg, 0, (size_t)N_NODES * HID * sizeof(float), stream);
    hipMemsetAsync(dxb, 0, (size_t)N_NODES * 3 * sizeof(float), stream);
    egnn_edge_kernel<<<edgeBlocks, 32 * WAVES_PER_BLOCK, 0, stream>>>(
        h_out, x_out, edge_attr, eidx, pe_b1, pe_b2, px_b1, px_w2, px_b2,
        wf, agg, dxb, l);
    egnn_node_kernel<<<nodeBlocks, 32 * WAVES_PER_BLOCK, 0, stream>>>(
        agg, dxb, ph_b1, ph_b2, ln_g, ln_b, wf, h_out, x_out, l);
  }
}
